// Seq2SeqPDESolver_64493228917100
// MI455X (gfx1250) — matrix operand order repaired and checked by reading, not compiled, not run on hardware
//
#include <hip/hip_runtime.h>
#include <hip/hip_bf16.h>
#include <cstdint>
#include <cstddef>

// Problem constants (match reference)
static constexpr int Bn   = 64;
static constexpr int Sn   = 256;
static constexpr int Tn   = 256;
static constexpr int DIN  = 128;
static constexpr int Hn   = 1024;
static constexpr int DOUT = 128;
static constexpr int Gn   = 4 * Hn;   // 4096 gate columns

typedef __attribute__((ext_vector_type(16))) __bf16 v16bf;
typedef __attribute__((ext_vector_type(8)))  __bf16 v8bf;
typedef __attribute__((ext_vector_type(8)))  float  v8f;
typedef __attribute__((ext_vector_type(4)))  unsigned int v4u;
typedef __attribute__((ext_vector_type(4)))  int v4i;

typedef __attribute__((address_space(1))) v4i gv4i;   // global <4 x i32>
typedef __attribute__((address_space(3))) v4i sv4i;   // LDS    <4 x i32>

__device__ __forceinline__ __bf16 f2bf(float f) {
    unsigned u = __builtin_bit_cast(unsigned, f);
    unsigned r = u + 0x7FFFu + ((u >> 16) & 1u);   // round-to-nearest-even
    unsigned short s = (unsigned short)(r >> 16);
    return __builtin_bit_cast(__bf16, s);
}

__device__ __forceinline__ float sigf(float x) {
    return 1.0f / (1.0f + __expf(-x));
}

// ---- CDNA5 async global->LDS staging (builtin present on this toolchain) ---
__device__ __forceinline__ void stage16(const __bf16* gp, __bf16* lp) {
#if __has_builtin(__builtin_amdgcn_global_load_async_to_lds_b128)
    __builtin_amdgcn_global_load_async_to_lds_b128(
        (gv4i*)(const_cast<__bf16*>(gp)), (sv4i*)(lp), 0, 0);
#else
    *(v4u*)lp = *(const v4u*)gp;
#endif
}
__device__ __forceinline__ void stage_wait() {
#if __has_builtin(__builtin_amdgcn_global_load_async_to_lds_b128)
 #if __has_builtin(__builtin_amdgcn_s_wait_asynccnt)
    __builtin_amdgcn_s_wait_asynccnt(0);
 #else
    asm volatile("s_wait_asynccnt 0x0" ::: "memory");
 #endif
#endif
}

// ---------------------------------------------------------------------------
// Fused LSTM step: gates GEMM (bf16 WMMA, f32 accum) + pointwise cell update.
// Each block owns a 32-wide h-column slice j0..j0+31 and computes the four
// gate tiles (i,f,g,o) for that slice: wave w -> gate g=w>>1, sub-tile (w&1).
// A tiles double-buffered via async DMA to LDS (fill of chunk i+1 overlaps
// WMMAs of chunk i; one barrier per chunk).  h ping-pongs (Hh -> hOut).
// ---------------------------------------------------------------------------
__global__ __launch_bounds__(256)
void lstm_step_fused(const __bf16* __restrict__ X,  long long ldX,
                     const __bf16* __restrict__ Wx, int Kx,
                     const __bf16* __restrict__ Hh,           // [64, Hn]
                     const __bf16* __restrict__ Wh,           // [Gn, Hn]
                     const float*  __restrict__ bias,         // [Gn]
                     float* __restrict__ c_st,                // [64, Hn] in-place
                     __bf16* __restrict__ hOut)               // [64, Hn]
{
    __shared__ __bf16 As[2][64][40];     // double-buffered 64x32 bf16 A tile
    __shared__ float  Gs[4][64][33];     // i,f,g,o gate tile for this j-slice

    const int t     = threadIdx.x;
    const int wave  = t >> 5;
    const int lane  = t & 31;
    const int j0    = blockIdx.x * 32;
    const int gate  = wave >> 1;
    const int n0    = gate * Hn + j0 + (wave & 1) * 16;
    const int col   = n0 + (lane & 15);     // global gate column (row of W)
    const int khalf = lane >> 4;
    const int klo   = khalf * 16;
    const int mrow  = lane & 15;
    const int srow  = t >> 2;
    const int sseg  = t & 3;

    const float bval = bias[col];
    v8f acc[4];
    #pragma unroll
    for (int mt = 0; mt < 4; ++mt)
        #pragma unroll
        for (int i = 0; i < 8; ++i) acc[mt][i] = bval;

    auto run_pass = [&](const __bf16* A, long long ldA, const __bf16* W, int K) {
        const int nch = K >> 5;                      // 32-wide K chunks
        // prologue: fill buffer 0
        stage16(A + (size_t)srow * ldA + sseg * 8, &As[0][srow][sseg * 8]);
        stage_wait();
        __syncthreads();
        for (int i = 0; i < nch; ++i) {
            const int k0 = i << 5;
            __bf16 (*cur)[40] = As[i & 1];
            if (i + 1 < nch)
                stage16(A + (size_t)srow * ldA + k0 + 32 + sseg * 8,
                        &As[(i + 1) & 1][srow][sseg * 8]);

            const __bf16* wp = W + (size_t)col * K + k0 + khalf * 16;
            v16bf bfrag = *(const v16bf*)wp;
            if (k0 + 32 < K)
                __builtin_prefetch((const void*)(wp + 32), 0, 3);

            // load all four A fragments first, then issue WMMAs back-to-back
            v16bf afr[4];
            #pragma unroll
            for (int mt = 0; mt < 4; ++mt) {
                const int m = mt * 16 + mrow;
                v8bf lo = *(const v8bf*)&cur[m][klo];
                v8bf hi = *(const v8bf*)&cur[m][klo + 8];
                afr[mt] = __builtin_shufflevector(lo, hi,
                    0,1,2,3,4,5,6,7,8,9,10,11,12,13,14,15);
            }
            #pragma unroll
            for (int mt = 0; mt < 4; ++mt)
                acc[mt] = __builtin_amdgcn_wmma_f32_16x16x32_bf16(
                    false, afr[mt], false, bfrag, (short)0, acc[mt], false, false);

            stage_wait();        // next-chunk DMA has completed by now (overlapped)
            __syncthreads();     // single barrier per chunk
        }
    };

    run_pass(X,  ldX,          Wx, Kx);   // input GEMM
    run_pass(Hh, (long long)Hn, Wh, Hn);  // recurrent GEMM

    // stash gate tile to LDS: D layout -> Gs[gate][m][jj]
    const int jj = (wave & 1) * 16 + (lane & 15);
    #pragma unroll
    for (int mt = 0; mt < 4; ++mt)
        #pragma unroll
        for (int r = 0; r < 8; ++r)
            Gs[gate][mt * 16 + r + 8 * khalf][jj] = acc[mt][r];
    __syncthreads();

    // pointwise LSTM cell update for the 64 x 32 slice
    for (int e = t; e < 64 * 32; e += 256) {
        const int m  = e >> 5;
        const int j2 = e & 31;
        const float gi = Gs[0][m][j2];
        const float gf = Gs[1][m][j2];
        const float gg = Gs[2][m][j2];
        const float go = Gs[3][m][j2];
        const size_t idx = (size_t)m * Hn + j0 + j2;
        const float cn = sigf(gf) * c_st[idx] + sigf(gi) * tanhf(gg);
        c_st[idx] = cn;
        hOut[idx] = f2bf(sigf(go) * tanhf(cn));
    }
}

// ---------------------------------------------------------------------------
// FC projection GEMM: out[64, N] = Hh@Wh^T + bias, plus bf16 copy (next input)
// ---------------------------------------------------------------------------
__global__ __launch_bounds__(256)
void fc_gemm_wmma(const __bf16* __restrict__ Hh, const __bf16* __restrict__ Wh, int K,
                  const float*  __restrict__ bias,
                  float* __restrict__ outF, long long ldOut,
                  __bf16* __restrict__ outBf, int ldBf)
{
    __shared__ __bf16 As[2][64][40];

    const int t     = threadIdx.x;
    const int wave  = t >> 5;
    const int lane  = t & 31;
    const int n0    = (blockIdx.x * 8 + wave) * 16;
    const int col   = n0 + (lane & 15);
    const int khalf = lane >> 4;
    const int klo   = khalf * 16;
    const int mrow  = lane & 15;
    const int srow  = t >> 2;
    const int sseg  = t & 3;

    const float bval = bias[col];
    v8f acc[4];
    #pragma unroll
    for (int mt = 0; mt < 4; ++mt)
        #pragma unroll
        for (int i = 0; i < 8; ++i) acc[mt][i] = bval;

    const int nch = K >> 5;
    stage16(Hh + (size_t)srow * K + sseg * 8, &As[0][srow][sseg * 8]);
    stage_wait();
    __syncthreads();
    for (int i = 0; i < nch; ++i) {
        const int k0 = i << 5;
        __bf16 (*cur)[40] = As[i & 1];
        if (i + 1 < nch)
            stage16(Hh + (size_t)srow * K + k0 + 32 + sseg * 8,
                    &As[(i + 1) & 1][srow][sseg * 8]);

        const __bf16* wp = Wh + (size_t)col * K + k0 + khalf * 16;
        v16bf bfrag = *(const v16bf*)wp;
        if (k0 + 32 < K)
            __builtin_prefetch((const void*)(wp + 32), 0, 3);

        v16bf afr[4];
        #pragma unroll
        for (int mt = 0; mt < 4; ++mt) {
            const int m = mt * 16 + mrow;
            v8bf lo = *(const v8bf*)&cur[m][klo];
            v8bf hi = *(const v8bf*)&cur[m][klo + 8];
            afr[mt] = __builtin_shufflevector(lo, hi,
                0,1,2,3,4,5,6,7,8,9,10,11,12,13,14,15);
        }
        #pragma unroll
        for (int mt = 0; mt < 4; ++mt)
            acc[mt] = __builtin_amdgcn_wmma_f32_16x16x32_bf16(
                false, afr[mt], false, bfrag, (short)0, acc[mt], false, false);

        stage_wait();
        __syncthreads();
    }

    #pragma unroll
    for (int mt = 0; mt < 4; ++mt)
        #pragma unroll
        for (int r = 0; r < 8; ++r) {
            const int m = mt * 16 + r + 8 * khalf;
            const float v = acc[mt][r];
            outF[(size_t)m * ldOut + col] = v;
            outBf[(size_t)m * ldBf + col] = f2bf(v);
        }
}

// f32 -> bf16 elementwise (grid-stride)
__global__ void cvt_f32_bf16(const float* __restrict__ in, __bf16* __restrict__ out, int n)
{
    for (int i = blockIdx.x * blockDim.x + threadIdx.x; i < n; i += gridDim.x * blockDim.x)
        out[i] = f2bf(in[i]);
}

__global__ void vec_add(const float* __restrict__ a, const float* __restrict__ b,
                        float* __restrict__ o, int n)
{
    int i = blockIdx.x * blockDim.x + threadIdx.x;
    if (i < n) o[i] = a[i] + b[i];
}

// x0_bf = bf16(trg[:, 0, :])
__global__ void cvt_trg0(const float* __restrict__ trg, __bf16* __restrict__ xbf)
{
    int i = blockIdx.x * blockDim.x + threadIdx.x;    // 64*128
    if (i < Bn * DOUT) {
        int m = i >> 7, n = i & 127;
        xbf[i] = f2bf(trg[(size_t)m * Tn * DOUT + n]);
    }
}

// out[:, 0, :] = 0
__global__ void zero_t0(float* __restrict__ out)
{
    int i = blockIdx.x * blockDim.x + threadIdx.x;    // 64*128
    if (i < Bn * DOUT) {
        int m = i >> 7, n = i & 127;
        out[(size_t)m * Tn * DOUT + n] = 0.0f;
    }
}

extern "C" void kernel_launch(void* const* d_in, const int* in_sizes, int n_in,
                              void* d_out, int out_size, void* d_ws, size_t ws_size,
                              hipStream_t stream)
{
    (void)in_sizes; (void)n_in; (void)out_size; (void)ws_size;

    const float* src     = (const float*)d_in[0];
    const float* trg     = (const float*)d_in[1];
    const float* enc_Wih = (const float*)d_in[2];
    const float* enc_Whh = (const float*)d_in[3];
    const float* enc_bih = (const float*)d_in[4];
    const float* enc_bhh = (const float*)d_in[5];
    const float* dec_Wih = (const float*)d_in[6];
    const float* dec_Whh = (const float*)d_in[7];
    const float* dec_bih = (const float*)d_in[8];
    const float* dec_bhh = (const float*)d_in[9];
    const float* fc_W    = (const float*)d_in[10];
    const float* fc_b    = (const float*)d_in[11];
    float* out = (float*)d_out;

    // workspace carve-up (256B aligned)
    char* p = (char*)d_ws;
    auto carve = [&](size_t bytes) { void* r = p; p += (bytes + 255) & ~(size_t)255; return r; };
    __bf16* encWih_bf = (__bf16*)carve((size_t)Gn * DIN  * 2);
    __bf16* encWhh_bf = (__bf16*)carve((size_t)Gn * Hn   * 2);
    __bf16* decWih_bf = (__bf16*)carve((size_t)Gn * DOUT * 2);
    __bf16* decWhh_bf = (__bf16*)carve((size_t)Gn * Hn   * 2);
    __bf16* fcW_bf    = (__bf16*)carve((size_t)DOUT * Hn * 2);
    __bf16* src_bf    = (__bf16*)carve((size_t)Bn * Sn * DIN * 2);
    float*  enc_b     = (float*) carve((size_t)Gn * 4);
    float*  dec_b     = (float*) carve((size_t)Gn * 4);
    float*  c_st      = (float*) carve((size_t)Bn * Hn * 4);
    __bf16* h0        = (__bf16*)carve((size_t)Bn * Hn * 2);
    __bf16* h1        = (__bf16*)carve((size_t)Bn * Hn * 2);
    __bf16* x_bf      = (__bf16*)carve((size_t)Bn * DOUT * 2);

    auto cvt = [&](const float* in, __bf16* o, int n) {
        int blocks = (n + 255) / 256; if (blocks > 4096) blocks = 4096;
        cvt_f32_bf16<<<blocks, 256, 0, stream>>>(in, o, n);
    };
    cvt(enc_Wih, encWih_bf, Gn * DIN);
    cvt(enc_Whh, encWhh_bf, Gn * Hn);
    cvt(dec_Wih, decWih_bf, Gn * DOUT);
    cvt(dec_Whh, decWhh_bf, Gn * Hn);
    cvt(fc_W,    fcW_bf,    DOUT * Hn);
    cvt(src,     src_bf,    Bn * Sn * DIN);
    vec_add<<<(Gn + 255) / 256, 256, 0, stream>>>(enc_bih, enc_bhh, enc_b, Gn);
    vec_add<<<(Gn + 255) / 256, 256, 0, stream>>>(dec_bih, dec_bhh, dec_b, Gn);

    (void)hipMemsetAsync(h0,   0, (size_t)Bn * Hn * 2, stream);
    (void)hipMemsetAsync(c_st, 0, (size_t)Bn * Hn * 4, stream);
    zero_t0<<<(Bn * DOUT + 255) / 256, 256, 0, stream>>>(out);

    __bf16* hin  = h0;
    __bf16* hout = h1;

    // ---------------- encoder: 256 fused steps ----------------
    for (int s = 0; s < Sn; ++s) {
        lstm_step_fused<<<Hn / 32, 256, 0, stream>>>(
            src_bf + (size_t)s * DIN, (long long)Sn * DIN, encWih_bf, DIN,
            hin, encWhh_bf, enc_b, c_st, hout);
        __bf16* tmp = hin; hin = hout; hout = tmp;
    }

    // ---------------- decoder: 255 fused steps + FC ----------------
    cvt_trg0<<<(Bn * DOUT + 255) / 256, 256, 0, stream>>>(trg, x_bf);
    for (int s = 0; s < Tn - 1; ++s) {
        lstm_step_fused<<<Hn / 32, 256, 0, stream>>>(
            x_bf, (long long)DOUT, decWih_bf, DOUT,
            hin, decWhh_bf, dec_b, c_st, hout);
        __bf16* tmp = hin; hin = hout; hout = tmp;
        // pred = h @ fc_W^T + fc_b -> out[:, s+1, :] and x_bf (next input)
        fc_gemm_wmma<<<DOUT / 128, 256, 0, stream>>>(
            hin, fcW_bf, Hn, fc_b,
            out + (size_t)(s + 1) * DOUT, (long long)Tn * DOUT,
            x_bf, DOUT);
    }
}
